// MultiHeadSelfAttention_20323785245243
// MI455X (gfx1250) — matrix operand order repaired and checked by reading, not compiled, not run on hardware
//
#include <hip/hip_runtime.h>
#include <hip/hip_bf16.h>

typedef __attribute__((ext_vector_type(16))) _Float16 v16h;
typedef __attribute__((ext_vector_type(8)))  float    v8f;

#define WMMA_F16(a, b, c) \
  __builtin_amdgcn_wmma_f32_16x16x32_f16(false, (a), false, (b), (short)0, (c), false, false)

// ---------------------------------------------------------------------------
// Fragment gather helpers (wave32; A/B use this code's contiguous k order, C as in the ISA)
// A 16x32 f16: lane = m + 16*(k/16);    half = k%16
// B 32x16 f16: lane = n + 16*(k/16);    half = k%16
// C 16x16 f32: lane = n + 16*(m/8);     vgpr = m%8
// ---------------------------------------------------------------------------

// A fragment from row-major f16 matrix (16 rows x 32 cols window), leading dim
// lda. Halves 0-7 and 8-15 are each 16B-contiguous -> 2x b128 loads per lane.
static __device__ __forceinline__ v16h load_a_f16(const _Float16* A, int lda) {
  const int lane = threadIdx.x & 31;
  const int m  = lane & 15;
  const int kg = (lane >> 4) & 1;           // k base 0 or 16
  const _Float16* row = A + (size_t)m * lda + kg * 16;
  v16h a;
#pragma unroll
  for (int i = 0; i < 8; ++i) { a[i] = row[i]; a[8 + i] = row[8 + i]; }
  return a;
}

// B fragment (32x16) where SOURCE is "column-contiguous": 16 rows of `ld` f16;
// row n supplies column n, the 16 k-values are contiguous (32B) per lane.
// Used for Q*K^T (rows of K), P*V (V transposed), and transposed f16 weights.
static __device__ __forceinline__ v16h load_bT_f16(const _Float16* Rows, int ld) {
  const int lane = threadIdx.x & 31;
  const int n  = lane & 15;
  const int kg = (lane >> 4) & 1;           // k base 0 or 16
  const _Float16* p = Rows + (size_t)n * ld + kg * 16;
  v16h b;
#pragma unroll
  for (int i = 0; i < 16; ++i) b[i] = p[i];
  return b;
}

// ---------------------------------------------------------------------------
// Constants
// ---------------------------------------------------------------------------
static constexpr int Bn = 4, T = 2048, C = 768, H = 12, D = 64;
static constexpr int BT = Bn * T;                // 8192

// ---------------------------------------------------------------------------
// Prep kernel A: convert X fp32 -> f16 (contiguous, fully coalesced).
// ---------------------------------------------------------------------------
__global__ __launch_bounds__(256) void cvt_x_kernel(
    const float* __restrict__ X, _Float16* __restrict__ Xh, int n) {
  const int i = blockIdx.x * 256 + threadIdx.x;
  if (i < n) Xh[i] = (_Float16)X[i];
}

// ---------------------------------------------------------------------------
// Prep kernel B: transpose + convert the four 768x768 weight matrices to f16.
// Wt[which][e][c] = (f16) W[c][e].  LDS-tiled 32x32, block = 256 (32x8).
// ---------------------------------------------------------------------------
__global__ __launch_bounds__(256) void transpose_w_kernel(
    const float* __restrict__ Wq, const float* __restrict__ Wk,
    const float* __restrict__ Wv, const float* __restrict__ Wo,
    _Float16* __restrict__ Wt) {
  __shared__ _Float16 tile[32][33];              // +1 pad to dodge bank conflicts
  const int which = blockIdx.z;
  const float* W = (which == 0) ? Wq : (which == 1) ? Wk : (which == 2) ? Wv : Wo;
  _Float16* dst = Wt + (size_t)which * C * C;
  const int e0 = blockIdx.x * 32, c0 = blockIdx.y * 32;
  const int tx = threadIdx.x & 31, ty = threadIdx.x >> 5;   // 32 x 8
#pragma unroll
  for (int i = 0; i < 32; i += 8)
    tile[ty + i][tx] = (_Float16)W[(size_t)(c0 + ty + i) * C + e0 + tx];
  __syncthreads();
#pragma unroll
  for (int i = 0; i < 32; i += 8)
    dst[(size_t)(e0 + ty + i) * C + c0 + tx] = tile[tx][ty + i];
}

// ---------------------------------------------------------------------------
// Kernel 1: fused QKV projection. One wave -> 16 (t) x 64 (one head) tile.
// grid = (BT/16, H, 3). A from f16 X, B from transposed f16 weights: all
// fragment loads contiguous. Q,K stored [B,H,T,D]; V transposed [B,H,D,T].
// ---------------------------------------------------------------------------
__global__ __launch_bounds__(32) void qkv_proj_kernel(
    const _Float16* __restrict__ Xh, const _Float16* __restrict__ Wt,
    _Float16* __restrict__ Qh, _Float16* __restrict__ Kh,
    _Float16* __restrict__ Vt) {
  const int bt0   = blockIdx.x * 16;
  const int h     = blockIdx.y;
  const int which = blockIdx.z;                  // 0=Q 1=K 2=V
  const _Float16* W = Wt + (size_t)which * C * C;   // transposed: W[e][c]
  const int ebase = h * D;

  v8f acc[4] = {};
  for (int k = 0; k < C; k += 32) {
    v16h a = load_a_f16(Xh + (size_t)bt0 * C + k, C);
#pragma unroll
    for (int j = 0; j < 4; ++j) {
      v16h b = load_bT_f16(W + (size_t)(ebase + j * 16) * C + k, C);
      acc[j] = WMMA_F16(a, b, acc[j]);
    }
  }

  const int lane = threadIdx.x & 31;
  const int g = lane >> 4, n16 = lane & 15;
#pragma unroll
  for (int j = 0; j < 4; ++j) {
#pragma unroll
    for (int r = 0; r < 8; ++r) {
      const int m  = r + 8 * g;
      const int tg = bt0 + m;
      const int b  = tg >> 11, tt = tg & (T - 1);
      const int d  = j * 16 + n16;
      const _Float16 val = (_Float16)acc[j][r];
      if (which == 2)
        Vt[(((size_t)b * H + h) * D + d) * T + tt] = val;          // [B,H,D,T]
      else {
        _Float16* dst = (which == 0) ? Qh : Kh;
        dst[(((size_t)b * H + h) * T + tt) * D + d] = val;         // [B,H,T,D]
      }
    }
  }
}

// ---------------------------------------------------------------------------
// Kernel 2: flash attention. One wave per (b,h, 16-query tile).
// Online softmax in registers (C-layout row stats), P bounced via LDS to
// convert C-layout -> A-fragment layout for the P*V WMMAs.
// ---------------------------------------------------------------------------
__global__ __launch_bounds__(32) void attn_kernel(
    const _Float16* __restrict__ Qh, const _Float16* __restrict__ Kh,
    const _Float16* __restrict__ Vt, _Float16* __restrict__ ctx) {
  __shared__ _Float16 Pbuf[16 * 32];

  const int tile  = blockIdx.x;                  // B*H*(T/16) tiles
  const int ttile = tile & (T / 16 - 1);
  const int bh    = tile >> 7;                   // T/16 == 128
  const int b     = bh / H, h = bh % H;
  const int t0    = ttile * 16;
  const float scale = 0.125f;                    // 1/sqrt(64)

  const _Float16* Qbase = Qh + ((size_t)bh * T + t0) * D;
  const _Float16* Kbase = Kh + (size_t)bh * T * D;
  const _Float16* Vbase = Vt + (size_t)bh * D * T;

  const v16h qa0 = load_a_f16(Qbase, D);         // Q cols 0..31
  const v16h qa1 = load_a_f16(Qbase + 32, D);    // Q cols 32..63

  const int lane = threadIdx.x & 31;
  const int g = lane >> 4, n16 = lane & 15;

  float run_max[8], run_sum[8];
  v8f O[4] = {};
#pragma unroll
  for (int r = 0; r < 8; ++r) { run_max[r] = -1e30f; run_sum[r] = 0.0f; }

  for (int s = 0; s < T; s += 32) {
    const v8f zero = {};
    // S = Q * K^T for 32 keys: two 16x16 tiles, K-dim 64 -> 2 WMMAs each.
    v16h bk = load_bT_f16(Kbase + (size_t)s * D, D);
    v8f s0  = WMMA_F16(qa0, bk, zero);
    bk      = load_bT_f16(Kbase + (size_t)s * D + 32, D);
    s0      = WMMA_F16(qa1, bk, s0);
    bk      = load_bT_f16(Kbase + (size_t)(s + 16) * D, D);
    v8f s1  = WMMA_F16(qa0, bk, zero);
    bk      = load_bT_f16(Kbase + (size_t)(s + 16) * D + 32, D);
    s1      = WMMA_F16(qa1, bk, s1);

    // Online softmax per row (row m = r + 8*g lives in vgpr r, lane group g).
#pragma unroll
    for (int r = 0; r < 8; ++r) {
      const float x0 = s0[r] * scale, x1 = s1[r] * scale;
      float mr = fmaxf(x0, x1);
      mr = fmaxf(mr, __shfl_xor(mr, 1));
      mr = fmaxf(mr, __shfl_xor(mr, 2));
      mr = fmaxf(mr, __shfl_xor(mr, 4));
      mr = fmaxf(mr, __shfl_xor(mr, 8));        // reduce within 16-lane group
      const float newmax = fmaxf(run_max[r], mr);
      const float corr   = __expf(run_max[r] - newmax);
      run_max[r] = newmax;
      const float p0 = __expf(x0 - newmax);
      const float p1 = __expf(x1 - newmax);
      float ps = p0 + p1;
      ps += __shfl_xor(ps, 1); ps += __shfl_xor(ps, 2);
      ps += __shfl_xor(ps, 4); ps += __shfl_xor(ps, 8);
      run_sum[r] = run_sum[r] * corr + ps;
#pragma unroll
      for (int j = 0; j < 4; ++j) O[j][r] *= corr;
      const int m = r + 8 * g;
      Pbuf[m * 32 + n16]      = (_Float16)p0;   // C-layout -> row-major LDS
      Pbuf[m * 32 + 16 + n16] = (_Float16)p1;
    }
    __syncthreads();
    const v16h pa = load_a_f16(Pbuf, 32);       // re-gather as A fragment
    __syncthreads();

    // O += P * V  (V transposed: contiguous B-fragment loads)
#pragma unroll
    for (int j = 0; j < 4; ++j) {
      v16h vb = load_bT_f16(Vbase + (size_t)(j * 16) * T + s, T);
      O[j] = WMMA_F16(pa, vb, O[j]);
    }
  }

  // Finalize: divide by running sum, write ctx [B,T,C] as f16.
#pragma unroll
  for (int r = 0; r < 8; ++r) {
    const float inv = 1.0f / run_sum[r];
    const int m = r + 8 * g;
    const int t = t0 + m;
#pragma unroll
    for (int j = 0; j < 4; ++j)
      ctx[((size_t)b * T + t) * C + h * D + j * 16 + n16] =
          (_Float16)(O[j][r] * inv);
  }
}

// ---------------------------------------------------------------------------
// Kernel 3: output projection  out = ctx @ Wo + bo   (fp32 out).
// B from transposed f16 Wo -> contiguous fragment loads.
// ---------------------------------------------------------------------------
__global__ __launch_bounds__(32) void out_proj_kernel(
    const _Float16* __restrict__ ctx, const _Float16* __restrict__ Wot,
    const float* __restrict__ bo, float* __restrict__ out) {
  const int bt0   = blockIdx.x * 16;
  const int ebase = blockIdx.y * 64;

  v8f acc[4] = {};
  for (int k = 0; k < C; k += 32) {
    v16h a = load_a_f16(ctx + (size_t)bt0 * C + k, C);
#pragma unroll
    for (int j = 0; j < 4; ++j) {
      v16h b = load_bT_f16(Wot + (size_t)(ebase + j * 16) * C + k, C);
      acc[j] = WMMA_F16(a, b, acc[j]);
    }
  }

  const int lane = threadIdx.x & 31;
  const int g = lane >> 4, n16 = lane & 15;
#pragma unroll
  for (int j = 0; j < 4; ++j) {
    const int e = ebase + j * 16 + n16;
    const float bias = bo[e];
#pragma unroll
    for (int r = 0; r < 8; ++r) {
      const int m = r + 8 * g;
      out[(size_t)(bt0 + m) * C + e] = acc[j][r] + bias;
    }
  }
}

// ---------------------------------------------------------------------------
// Launch. Workspace layout (f16 elements):
//   Xh : BT*C = 6,291,456 | Wt : 4*C*C = 2,359,296 | Qh,Kh,Vt,ctx : 6,291,456 each
// Total ~67.6 MB.
// ---------------------------------------------------------------------------
extern "C" void kernel_launch(void* const* d_in, const int* in_sizes, int n_in,
                              void* d_out, int out_size, void* d_ws, size_t ws_size,
                              hipStream_t stream) {
  const float* X  = (const float*)d_in[0];
  const float* Wq = (const float*)d_in[1];
  const float* Wk = (const float*)d_in[2];
  const float* Wv = (const float*)d_in[3];
  const float* Wo = (const float*)d_in[4];
  const float* bo = (const float*)d_in[5];
  float* out = (float*)d_out;

  const size_t NX = (size_t)BT * C;              // 6,291,456
  const size_t NW = (size_t)C * C;               //   589,824
  _Float16* Xh  = (_Float16*)d_ws;
  _Float16* Wt  = Xh + NX;                       // 4 transposed matrices
  _Float16* Qh  = Wt + 4 * NW;
  _Float16* Kh  = Qh + NX;
  _Float16* Vt  = Kh + NX;
  _Float16* ctx = Vt + NX;
  _Float16* Wot = Wt + 3 * NW;                   // Wo^T

  cvt_x_kernel<<<dim3((int)(NX / 256)), 256, 0, stream>>>(X, Xh, (int)NX);
  transpose_w_kernel<<<dim3(C / 32, C / 32, 4), 256, 0, stream>>>(Wq, Wk, Wv, Wo, Wt);
  qkv_proj_kernel<<<dim3(BT / 16, H, 3), 32, 0, stream>>>(Xh, Wt, Qh, Kh, Vt);
  attn_kernel<<<dim3(Bn * H * (T / 16)), 32, 0, stream>>>(Qh, Kh, Vt, ctx);
  out_proj_kernel<<<dim3(BT / 16, C / 64), 32, 0, stream>>>(ctx, Wot, bo, out);
}
